// PointwiseAttention_65377992179728
// MI455X (gfx1250) — matrix operand order repaired and checked by reading, not compiled, not run on hardware
//
#include <hip/hip_runtime.h>

// ---------- types ----------
typedef _Float16 f16;
typedef __attribute__((ext_vector_type(16))) _Float16 v16h;
typedef __attribute__((ext_vector_type(8)))  _Float16 v8h;
typedef __attribute__((ext_vector_type(8)))  float    v8f;

typedef int i32x4 __attribute__((vector_size(16)));
typedef __attribute__((address_space(1))) i32x4* as1_b128;
typedef __attribute__((address_space(3))) i32x4* as3_b128;
typedef __attribute__((address_space(1))) int*   as1_b32;
typedef __attribute__((address_space(3))) int*   as3_b32;

// global -> LDS async copies (gfx1250, tracked by ASYNCcnt)
__device__ __forceinline__ void async_copy_b128(const void* g, void* l) {
    __builtin_amdgcn_global_load_async_to_lds_b128(
        (as1_b128)(size_t)g, (as3_b128)(unsigned int)(size_t)l, 0, 0);
}
__device__ __forceinline__ void async_copy_b32(const void* g, void* l) {
    __builtin_amdgcn_global_load_async_to_lds_b32(
        (as1_b32)(size_t)g, (as3_b32)(unsigned int)(size_t)l, 0, 0);
}
__device__ __forceinline__ void wait_async0() {
    asm volatile("s_wait_asynccnt 0x0" ::: "memory");
}

// ---------- problem constants ----------
#define BATCH   2
#define QLEN    4096
#define KLEN    4096
#define KEYDIM  128
#define VALDIM  64
#define NHEAD   4
#define HK      32      // head key dim  == one WMMA K step (f16 16x16x32)
#define HV      16      // head value dim == one WMMA N block
#define OUTDIM  64
#define QTILE   128     // q rows per workgroup: 8 wave32 x 16 rows
#define KTILE   64      // kv rows per inner tile (4 col-blocks of 16)
#define NWAVE   8
#define NKT     (KLEN / KTILE)

// =====================================================================
// Kernel 1: Q projection  (B*Q blocks x 128 thr)  Qp[b][h][q][32] f16
// =====================================================================
__global__ void __launch_bounds__(128)
proj_q_kernel(const float* __restrict__ qd, const float* __restrict__ qw,
              f16* __restrict__ Qp)
{
    __shared__ float row[KEYDIM];
    const int bq = blockIdx.x;
    const int b = bq / QLEN, q = bq % QLEN;
    const int t = threadIdx.x;                 // t = h*32 + c
    row[t] = qd[(size_t)bq * KEYDIM + t];
    __syncthreads();
    float acc = 0.f;
    #pragma unroll 8
    for (int a = 0; a < KEYDIM; ++a)
        acc += row[a] * qw[a * (NHEAD * HK) + t];
    const int h = t >> 5, c = t & 31;
    const float scale = 0.17677669529663687f;  // 1/sqrt(32)
    Qp[(((size_t)(b * NHEAD + h) * QLEN) + q) * HK + c] = (f16)(acc * scale);
}

// =====================================================================
// Kernel 2: K/V projections (B*K blocks x 192 thr)
//   Kp[b][h][k][32] f16,  Vp[b][h][k][16] f16
// =====================================================================
__global__ void __launch_bounds__(192)
proj_kv_kernel(const float* __restrict__ md, const float* __restrict__ kw,
               const float* __restrict__ vw, f16* __restrict__ Kp,
               f16* __restrict__ Vp)
{
    __shared__ float row[VALDIM];
    const int bk = blockIdx.x;
    const int b = bk / KLEN, k = bk % KLEN;
    const int t = threadIdx.x;
    if (t < VALDIM) row[t] = md[(size_t)bk * VALDIM + t];
    __syncthreads();
    if (t < NHEAD * HK) {                       // K projection, t = h*32+c
        float acc = 0.f;
        #pragma unroll 8
        for (int a = 0; a < VALDIM; ++a)
            acc += row[a] * kw[a * (NHEAD * HK) + t];
        const int h = t >> 5, c = t & 31;
        Kp[(((size_t)(b * NHEAD + h) * KLEN) + k) * HK + c] = (f16)acc;
    } else {                                    // V projection, t2 = h*16+c
        const int t2 = t - NHEAD * HK;
        float acc = 0.f;
        #pragma unroll 8
        for (int a = 0; a < VALDIM; ++a)
            acc += row[a] * vw[a * (NHEAD * HV) + t2];
        const int h = t2 >> 4, c = t2 & 15;
        Vp[(((size_t)(b * NHEAD + h) * KLEN) + k) * HV + c] = (f16)acc;
    }
}

// =====================================================================
// Kernel 3: fused flash attention core (WMMA f16 -> f32)
//   grid = (QLEN/QTILE, NHEAD, BATCH), 256 threads = 8 wave32
//   Double-buffered pipeline per 64-row KV tile:
//     tile kt+1's K/mask DMA (GLOBAL_LOAD_ASYNC_TO_LDS) runs while
//     tile kt's 6 WMMAs + online softmax execute.
// =====================================================================
__global__ void __launch_bounds__(256)
fa_kernel(const f16* __restrict__ Qp, const f16* __restrict__ Kp,
          const f16* __restrict__ Vp, const int* __restrict__ mask,
          float* __restrict__ wa)
{
    __shared__ __align__(32) f16 Kt[2][KTILE][HK];     // K tiles (async DMA)
    __shared__ __align__(16) f16 Vt[2][HV][KTILE + 8]; // V tiles transposed
    __shared__ int Msk[2][KTILE];
    __shared__ __align__(16) f16 Pbuf[NWAVE][16][72];  // per-wave P staging

    const int qblk = blockIdx.x, h = blockIdx.y, b = blockIdx.z;
    const int wave = threadIdx.x >> 5;
    const int lane = threadIdx.x & 31;
    const int lcol = lane & 15;              // N column / row-in-half
    const int lhalf = lane >> 4;             // 0: lanes 0-15, 1: lanes 16-31

    const f16* Qph = Qp + (size_t)(b * NHEAD + h) * QLEN * HK;
    const f16* Kph = Kp + (size_t)(b * NHEAD + h) * KLEN * HK;
    const f16* Vph = Vp + (size_t)(b * NHEAD + h) * KLEN * HV;
    const int* mb  = mask + (size_t)b * KLEN;

    const int q0 = qblk * QTILE + wave * 16;

    auto issue_dma = [&](int buf, int kbase) {
        const f16* gk = Kph + (size_t)kbase * HK + threadIdx.x * 8;
        async_copy_b128(gk, &Kt[buf][0][0] + threadIdx.x * 8);
        if (threadIdx.x < KTILE)
            async_copy_b32(mb + kbase + threadIdx.x, &Msk[buf][threadIdx.x]);
    };

    // --- Q A-fragment (16x32 f16), ISA 7.12.2 layout, held for whole loop ---
    v16h qf;
    {
        const f16* rp = Qph + (size_t)(q0 + lcol) * HK;
        const v8h lo = *(const v8h*)(rp + lhalf * 8);        // K 0-7 / 8-15
        const v8h hi = *(const v8h*)(rp + 16 + lhalf * 8);   // K 16-23 / 24-31
        #pragma unroll
        for (int i = 0; i < 8; ++i) { qf[i] = lo[i]; qf[i + 8] = hi[i]; }
    }

    v8f acc = {};
    float m[8], l[8];
    #pragma unroll
    for (int r = 0; r < 8; ++r) { m[r] = -1e30f; l[r] = 0.f; }

    issue_dma(0, 0);                         // prologue: tile 0 in flight

    for (int kt = 0; kt < NKT; ++kt) {
        const int cur = kt & 1;
        const int kbase = kt * KTILE;

        // ---- stage V^T for current tile via registers ----
        {
            const int r = threadIdx.x >> 2;            // 0..63 kv row
            const int g = (threadIdx.x & 3) * 4;       // 4-col group
            const f16* vp = Vph + (size_t)(kbase + r) * HV + g;
            const f16 v0 = vp[0], v1 = vp[1], v2 = vp[2], v3 = vp[3];
            Vt[cur][g + 0][r] = v0; Vt[cur][g + 1][r] = v1;
            Vt[cur][g + 2][r] = v2; Vt[cur][g + 3][r] = v3;
        }
        wait_async0();                       // current tile's DMA has landed
        __syncthreads();                     // tile visible; old-buffer readers done

        if (kt + 1 < NKT) issue_dma(cur ^ 1, kbase + KTILE);  // overlap next DMA

        // ---- S = Q x K^T : 4 column blocks, one wmma each (B from LDS) ----
        v8f S[4];
        #pragma unroll
        for (int n = 0; n < 4; ++n) {
            // B-matrix (32x16): lane = kv column, elems = head dims
            const f16* kp = &Kt[cur][n * 16 + lcol][0] + lhalf * 8;
            const v8h klo = *(const v8h*)kp;
            const v8h khi = *(const v8h*)(kp + 16);
            v16h kf;
            #pragma unroll
            for (int i = 0; i < 8; ++i) { kf[i] = klo[i]; kf[i + 8] = khi[i]; }
            v8f z = {};
            S[n] = __builtin_amdgcn_wmma_f32_16x16x32_f16(
                       false, qf, false, kf, (short)0, z, false, false);
        }

        // ---- mask (branch-free; col = kbase + n*16 + lcol) ----
        #pragma unroll
        for (int n = 0; n < 4; ++n) {
            const bool dead = (Msk[cur][n * 16 + lcol] == 0);
            #pragma unroll
            for (int r = 0; r < 8; ++r)
                S[n][r] = dead ? -1e30f : S[n][r];
        }

        // ---- online softmax: row max across 16 lanes of the half ----
        float mt[8];
        #pragma unroll
        for (int r = 0; r < 8; ++r) {
            float v = fmaxf(fmaxf(S[0][r], S[1][r]), fmaxf(S[2][r], S[3][r]));
            v = fmaxf(v, __shfl_xor(v, 1, 32));
            v = fmaxf(v, __shfl_xor(v, 2, 32));
            v = fmaxf(v, __shfl_xor(v, 4, 32));
            v = fmaxf(v, __shfl_xor(v, 8, 32));
            mt[r] = v;
        }
        #pragma unroll
        for (int r = 0; r < 8; ++r) {
            const float mn = fmaxf(m[r], mt[r]);
            const float sc = __expf(m[r] - mn);
            m[r] = mn; l[r] *= sc; acc[r] *= sc;
        }

        // ---- P = exp(S - m); stage to per-wave LDS in row-major ----
        #pragma unroll
        for (int n = 0; n < 4; ++n) {
            #pragma unroll
            for (int r = 0; r < 8; ++r) {
                const float p = __expf(S[n][r] - m[r]);
                S[n][r] = p;
                Pbuf[wave][r + 8 * lhalf][n * 16 + lcol] = (f16)p;
            }
        }
        #pragma unroll
        for (int r = 0; r < 8; ++r) {
            float s = S[0][r] + S[1][r] + S[2][r] + S[3][r];
            s += __shfl_xor(s, 1, 32);
            s += __shfl_xor(s, 2, 32);
            s += __shfl_xor(s, 4, 32);
            s += __shfl_xor(s, 8, 32);
            l[r] += s;
        }

        // ---- O += P x V : 2 K-chunks of 32 (same-wave LDS is in-order) ----
        #pragma unroll
        for (int c = 0; c < 2; ++c) {
            // P A-fragment: row = lcol, K halves per ISA layout
            const f16* pr = &Pbuf[wave][lcol][0];
            const v8h plo = *(const v8h*)(pr + c * 32 + lhalf * 8);
            const v8h phi = *(const v8h*)(pr + c * 32 + 16 + lhalf * 8);
            v16h pf;
            #pragma unroll
            for (int i = 0; i < 8; ++i) { pf[i] = plo[i]; pf[i + 8] = phi[i]; }
            // V B-fragment: lane = vdim column, elems = kv index
            const f16* vr = &Vt[cur][lcol][c * 32 + lhalf * 8];
            const v8h vlo = *(const v8h*)vr;
            const v8h vhi = *(const v8h*)(vr + 16);
            v16h vf;
            #pragma unroll
            for (int i = 0; i < 8; ++i) { vf[i] = vlo[i]; vf[i + 8] = vhi[i]; }
            acc = __builtin_amdgcn_wmma_f32_16x16x32_f16(
                      false, pf, false, vf, (short)0, acc, false, false);
        }
        __syncthreads();                     // done with 'cur' before re-DMA at kt+2
    }

    // ---- finalize: O /= l, store weight_avg [b][q][h*16+c] f32 ----
    float* wap = wa + (size_t)b * QLEN * (NHEAD * HV);
    #pragma unroll
    for (int r = 0; r < 8; ++r) {
        const float inv = 1.0f / l[r];
        const int q = q0 + r + 8 * lhalf;
        wap[(size_t)q * (NHEAD * HV) + h * HV + lcol] = acc[r] * inv;
    }
}

// =====================================================================
// Kernel 4: output projection + bias (f32 for accuracy; K=64 is cheap)
// =====================================================================
__global__ void __launch_bounds__(256)
oproj_kernel(const float* __restrict__ wa, const float* __restrict__ ow,
             const float* __restrict__ ob, float* __restrict__ out)
{
    const int t = threadIdx.x;
    const int o = t & 63, ry = t >> 6;
    const size_t row = (size_t)blockIdx.x * 4 + ry;   // b*QLEN + q
    const float* wr = wa + row * (NHEAD * HV);
    float acc = ob[o];
    #pragma unroll
    for (int j = 0; j < NHEAD * HV; ++j)
        acc += wr[j] * ow[j * OUTDIM + o];
    out[row * OUTDIM + o] = acc;
}

// =====================================================================
extern "C" void kernel_launch(void* const* d_in, const int* in_sizes, int n_in,
                              void* d_out, int out_size, void* d_ws, size_t ws_size,
                              hipStream_t stream)
{
    const float* qd   = (const float*)d_in[0];
    const float* md   = (const float*)d_in[1];
    const int*   mask = (const int*)  d_in[2];
    const float* qw   = (const float*)d_in[3];
    const float* kw   = (const float*)d_in[4];
    const float* vw   = (const float*)d_in[5];
    const float* ow   = (const float*)d_in[6];
    const float* ob   = (const float*)d_in[7];
    float* out = (float*)d_out;

    char* ws = (char*)d_ws;
    f16*   Qp = (f16*)(ws);                         // 2 MB
    f16*   Kp = (f16*)(ws + (size_t)(2u << 20));    // 2 MB
    f16*   Vp = (f16*)(ws + (size_t)(4u << 20));    // 1 MB
    float* wa = (float*)(ws + (size_t)(5u << 20));  // 2 MB

    proj_q_kernel <<<BATCH * QLEN, 128, 0, stream>>>(qd, qw, Qp);
    proj_kv_kernel<<<BATCH * KLEN, 192, 0, stream>>>(md, kw, vw, Kp, Vp);
    fa_kernel<<<dim3(QLEN / QTILE, NHEAD, BATCH), 256, 0, stream>>>(Qp, Kp, Vp, mask, wa);
    oproj_kernel<<<(BATCH * QLEN) / 4, 256, 0, stream>>>(wa, ow, ob, out);
}
